// BahdanauGRUCell_43851616092365
// MI455X (gfx1250) — matrix operand order repaired and checked by reading, not compiled, not run on hardware
//
#include <hip/hip_runtime.h>
#include <hip/hip_bf16.h>
#include <math.h>

// ---------------------------------------------------------------------------
// Bahdanau GRU cell, MI455X (gfx1250, wave32).
// Dominant work: annotations(65536x512) @ Ua(512x512) = 34.4 GFLOP fused with
// tanh()*Va reduction -> scores, on v_wmma_f32_16x16x32_bf16.
// Ua is pre-converted to a transposed bf16 copy once (512 KB), then streamed
// into LDS with the Tensor Data Mover (tensor_load_to_lds, double-buffered,
// s_wait_tensorcnt) -- the D# pad fields reproduce the LDS row padding.
// annotations (128 MB fp32, ~5.5us HBM floor) staged with float4 loads +
// in-register bf16 conversion.
// ---------------------------------------------------------------------------

#define Bb  32
#define Tt  2048
#define Ff  512
#define Uu  512
#define U3  1536

typedef __attribute__((ext_vector_type(16))) __bf16 v16bf;
typedef __attribute__((ext_vector_type(8)))  __bf16 v8bf;
typedef __attribute__((ext_vector_type(8)))  float  v8f;
typedef __attribute__((ext_vector_type(4)))  unsigned int u32x4;
typedef __attribute__((ext_vector_type(8)))  int i32x8;
typedef __attribute__((ext_vector_type(4)))  int i32x4;

__device__ __forceinline__ unsigned short f2bf(float f) {
    unsigned u = __float_as_uint(f);
    unsigned r = u + 0x7fffu + ((u >> 16) & 1u);   // RNE fp32 -> bf16
    return (unsigned short)(r >> 16);
}

__device__ __forceinline__ float hsig(float x) {
    return fminf(fmaxf(0.2f * x + 0.5f, 0.0f), 1.0f);
}

// ---------------------------------------------------------------------------
// K0: UaT[n][k] = bf16(Ua[k][n])  -- one-off 1 MB transpose+convert so the
// hot kernel can stage B-slabs with the TDM (no per-block VGPR conversion).
// ---------------------------------------------------------------------------
__global__ __launch_bounds__(256) void ua_transpose_bf16(
    const float* __restrict__ Ua, unsigned short* __restrict__ UaT)
{
    __shared__ float tile[32][33];
    const int tx = threadIdx.x & 31, ty = threadIdx.x >> 5;
    const int kb = blockIdx.y * 32, nb = blockIdx.x * 32;
    for (int i = ty; i < 32; i += 8)
        tile[i][tx] = Ua[(size_t)(kb + i) * Uu + nb + tx];
    __syncthreads();
    for (int i = ty; i < 32; i += 8)
        UaT[(size_t)(nb + i) * Ff + kb + tx] = f2bf(tile[tx][i]);
}

// ---------------------------------------------------------------------------
// K1: small fp32 GEMMs: hwa = h@Wa + Wa_bias + Ua_bias   (32x512)
//                       xm  = x@kernel + bias            (32x1536)
//                       hrec= h@recurrent_kernel         (32x1536)
// ---------------------------------------------------------------------------
__global__ __launch_bounds__(256) void small_mm(
    const float* __restrict__ x, const float* __restrict__ h,
    const float* __restrict__ kern, const float* __restrict__ rker,
    const float* __restrict__ Wa, const float* __restrict__ bias,
    const float* __restrict__ Wa_bias, const float* __restrict__ Ua_bias,
    float* __restrict__ hwa, float* __restrict__ xm, float* __restrict__ hrec)
{
    int gid = blockIdx.x * 256 + threadIdx.x;
    if (gid >= Bb * (Uu + U3 + U3)) return;
    int b = gid / (Uu + U3 + U3);
    int j = gid - b * (Uu + U3 + U3);
    if (j < Uu) {
        float a = Wa_bias[j] + Ua_bias[j];
        for (int k = 0; k < Uu; ++k) a += h[b * Uu + k] * Wa[(size_t)k * Uu + j];
        hwa[b * Uu + j] = a;
    } else if (j < Uu + U3) {
        int jj = j - Uu;
        float a = bias[jj];
        for (int k = 0; k < Ff; ++k) a += x[b * Ff + k] * kern[(size_t)k * U3 + jj];
        xm[b * U3 + jj] = a;
    } else {
        int jj = j - Uu - U3;
        float a = 0.f;
        for (int k = 0; k < Uu; ++k) a += h[b * Uu + k] * rker[(size_t)k * U3 + jj];
        hrec[b * U3 + jj] = a;
    }
}

// ---------------------------------------------------------------------------
// K2: scores[b,t] += sum_n tanh( (ann @ Ua)[t,n] + hwa[b,n] ) * Va[n]
// Tile: 16 t-rows x 256 N-cols; K streamed as 32-wide bf16 slabs via TDM
// (double-buffered). 8 waves x 2 N-tiles; 2 WMMAs / wave / K-step.
// ---------------------------------------------------------------------------
#define A_ROWS   16
#define A_STRIDE 520   // ushort; 1040B rows: 16B aligned, bank-rotating
#define BN       256   // N columns per block
#define B_STRIDE 40    // ushort; 80B rows = 32 bf16 data + 8 pad (TDM pad)
#define KSLAB    32
#define NSTEPS   (Ff / KSLAB)

// Issue one TDM slab load: UaT tile [BN rows x 32 cols] -> LDS, padded rows.
// D# fields per ISA 8.3/8.4: data_size=2B, pad after every 16 DWORDs (one
// 32-bf16 row) by 4 DWORDs (8 bf16) -> LDS row stride 40 ushorts.
__device__ __forceinline__ void tdm_load_slab(const unsigned short* gsrc,
                                              unsigned lds_byte_off)
{
    unsigned long long ga = (unsigned long long)(uintptr_t)gsrc;
    u32x4 g0 = {
        0x1u,                                   // count=1, user descriptor
        lds_byte_off,                           // lds_addr
        (unsigned)(ga & 0xffffffffu),           // global_addr[31:0]
        (unsigned)((ga >> 32) & 0x01ffffffu) | 0x80000000u  // [56:32]|type=2
    };
    i32x8 g1;
    g1[0] = (1 << 16) | (1 << 20) | (3 << 22) | (3 << 25);
    //       data_size=2B | pad_en | pad_interval=16dw | pad_amount=4dw
    g1[1] = (int)(512u << 16);   // tensor_dim0 = 512 (lo16)
    g1[2] = (int)(512u << 16);   // tensor_dim0 hi=0 | tensor_dim1 = 512 (lo16)
    g1[3] = (int)((unsigned)KSLAB << 16);  // tensor_dim1 hi=0 | tile_dim0 = 32
    g1[4] = BN;                  // tile_dim1 = 256, tile_dim2 = 0
    g1[5] = Ff;                  // tensor_dim0_stride = 512 (lo32)
    g1[6] = 0;                   // stride hi | tensor_dim1_stride lo (unused)
    g1[7] = 0;
    i32x4 z4 = (i32x4)0;
#if __clang_major__ >= 23
    __builtin_amdgcn_tensor_load_to_lds(g0, g1, z4, z4, (i32x8)0, 0);
#else
    __builtin_amdgcn_tensor_load_to_lds(g0, g1, z4, z4, 0);
#endif
}

__global__ __launch_bounds__(256) void attn_scores_wmma(
    const float* __restrict__ ann,            // [B][T][F] fp32
    const unsigned short* __restrict__ UaT,   // [U][F] bf16 (transposed)
    const float* __restrict__ hwa,            // [B][U] (biases folded)
    const float* __restrict__ Va,             // [U]
    float* __restrict__ scores)               // [B][T] pre-zeroed, atomics
{
    __shared__ __align__(16) unsigned short As[A_ROWS * A_STRIDE]; // 16,640 B
    __shared__ __align__(16) unsigned short Bs[2][BN * B_STRIDE];  // 40,960 B
    __shared__ float sHwa[BN];
    __shared__ float sVa[BN];
    __shared__ float sScore[A_ROWS];

    const int tid = threadIdx.x;
    const int b   = blockIdx.y;
    const int t0  = blockIdx.x * A_ROWS;
    const int n0  = blockIdx.z * BN;

    sHwa[tid] = hwa[b * Uu + n0 + tid];
    sVa[tid]  = Va[n0 + tid];
    if (tid < A_ROWS) sScore[tid] = 0.f;

    const int lane = tid & 31;
    const int l15  = lane & 15;
    const int hi   = lane >> 4;
    const int wave = tid >> 5;

    // Kick off TDM for slab 0 while we stage A (overlap DMA with conversion).
    if (wave == 0)
        tdm_load_slab(UaT + (size_t)n0 * Ff, (unsigned)(uintptr_t)&Bs[0][0]);

    // Stage annotations tile 16x512 fp32 -> bf16 LDS: tile is one contiguous
    // 8192-float span; float4 loads (global_load_b128) + 8B packed LDS stores.
    const float4* ab4 = (const float4*)(ann + ((size_t)b * Tt + t0) * Ff);
    for (int e = tid; e < A_ROWS * Ff / 4; e += 256) {
        float4 v = ab4[e];
        unsigned long long p =
            (unsigned long long)f2bf(v.x) |
            ((unsigned long long)f2bf(v.y) << 16) |
            ((unsigned long long)f2bf(v.z) << 32) |
            ((unsigned long long)f2bf(v.w) << 48);
        *(unsigned long long*)&As[(e >> 7) * A_STRIDE + (e & 127) * 4] = p;
    }

    v8f zero = {0.f, 0.f, 0.f, 0.f, 0.f, 0.f, 0.f, 0.f};
    v8f acc[2] = {zero, zero};
    const int ntb = wave * 2;   // this wave's 2 N-tiles

    for (int s = 0; s < NSTEPS; ++s) {
        if (wave == 0) {
            if (s + 1 < NSTEPS) {
                tdm_load_slab(UaT + (size_t)n0 * Ff + (s + 1) * KSLAB,
                              (unsigned)(uintptr_t)&Bs[(s + 1) & 1][0]);
                __builtin_amdgcn_s_wait_tensorcnt(1);  // slab s done (in-order)
            } else {
                __builtin_amdgcn_s_wait_tensorcnt(0);
            }
        }
        __syncthreads();   // slab s visible to all waves

        const unsigned short* bsb = &Bs[s & 1][0];
        // A frag: lanes 0-15 K[0,16), lanes 16-31 K[16,32)
        const unsigned short* ap = &As[l15 * A_STRIDE + s * KSLAB + hi * 16];
        v8bf alo = *reinterpret_cast<const v8bf*>(ap);
        v8bf ahi = *reinterpret_cast<const v8bf*>(ap + 8);
        v16bf afrag = __builtin_shufflevector(alo, ahi,
            0, 1, 2, 3, 4, 5, 6, 7, 8, 9, 10, 11, 12, 13, 14, 15);
        // B frags: lane = col n, lanes 0-15 K[0,16), lanes 16-31 K[16,32)
        const unsigned short* bp0 = &bsb[((ntb)     * 16 + l15) * B_STRIDE + hi * 16];
        const unsigned short* bp1 = &bsb[((ntb + 1) * 16 + l15) * B_STRIDE + hi * 16];
        v8bf b00 = *reinterpret_cast<const v8bf*>(bp0);
        v8bf b01 = *reinterpret_cast<const v8bf*>(bp0 + 8);
        v8bf b10 = *reinterpret_cast<const v8bf*>(bp1);
        v8bf b11 = *reinterpret_cast<const v8bf*>(bp1 + 8);
        v16bf bf0 = __builtin_shufflevector(b00, b01,
            0, 1, 2, 3, 4, 5, 6, 7, 8, 9, 10, 11, 12, 13, 14, 15);
        v16bf bf1 = __builtin_shufflevector(b10, b11,
            0, 1, 2, 3, 4, 5, 6, 7, 8, 9, 10, 11, 12, 13, 14, 15);
        acc[0] = __builtin_amdgcn_wmma_f32_16x16x32_bf16(
            false, afrag, false, bf0, (short)0, acc[0], false, false);
        acc[1] = __builtin_amdgcn_wmma_f32_16x16x32_bf16(
            false, afrag, false, bf1, (short)0, acc[1], false, false);
        __syncthreads();   // buf (s&1) fully consumed before slab s+2 lands
    }

    // Epilogue: sum_n tanh(acc + hwa[n]) * Va[n], C layout:
    // VGPR r: lanes 0-15 -> (M=r, N=l15), lanes 16-31 -> (M=r+8).
    float sv[8];
#pragma unroll
    for (int r = 0; r < 8; ++r) sv[r] = 0.f;
#pragma unroll
    for (int nt = 0; nt < 2; ++nt) {
        int n = (ntb + nt) * 16 + l15;
        float hb = sHwa[n], w = sVa[n];
#pragma unroll
        for (int r = 0; r < 8; ++r)
            sv[r] += tanhf(acc[nt][r] + hb) * w;
    }
#pragma unroll
    for (int r = 0; r < 8; ++r) {
        float v = sv[r];
        v += __shfl_xor(v, 1, 16);
        v += __shfl_xor(v, 2, 16);
        v += __shfl_xor(v, 4, 16);
        v += __shfl_xor(v, 8, 16);
        if (l15 == 0)
            atomicAdd(&sScore[hi * 8 + r], v);          // ds_add_f32
    }
    __syncthreads();
    if (tid < A_ROWS)   // combine the 2 N-half blocks
        atomicAdd(&scores[(size_t)b * Tt + t0 + tid], sScore[tid]);
}

// ---------------------------------------------------------------------------
// K3: softmax over T + context  c[b,f] = sum_t alpha[t] * ann[b,t,f]
// ---------------------------------------------------------------------------
__global__ __launch_bounds__(256) void softmax_ctx(
    const float* __restrict__ scores, const float* __restrict__ ann,
    float* __restrict__ c)
{
    __shared__ float alpha[Tt];
    __shared__ float red[256];
    const int b = blockIdx.x, tid = threadIdx.x;
    const float* sc = scores + (size_t)b * Tt;

    float mx = -1e30f;
    for (int t = tid; t < Tt; t += 256) mx = fmaxf(mx, sc[t]);
    red[tid] = mx; __syncthreads();
    for (int s = 128; s; s >>= 1) {
        if (tid < s) red[tid] = fmaxf(red[tid], red[tid + s]);
        __syncthreads();
    }
    mx = red[0]; __syncthreads();

    float sum = 0.f;
    for (int t = tid; t < Tt; t += 256) {
        float e = __expf(sc[t] - mx);
        alpha[t] = e;
        sum += e;
    }
    red[tid] = sum; __syncthreads();
    for (int s = 128; s; s >>= 1) {
        if (tid < s) red[tid] += red[tid + s];
        __syncthreads();
    }
    const float inv = 1.f / red[0];

    const int f = blockIdx.y * 256 + tid;
    const float* abp = ann + (size_t)b * Tt * Ff + f;
    float a = 0.f;
    for (int t = 0; t < Tt; ++t) a += alpha[t] * abp[(size_t)t * Ff];
    c[b * Uu + f] = a * inv;
}

// ---------------------------------------------------------------------------
// K4: am = c @ attention_kernel + attention_bias, fused GRU gate update.
// ---------------------------------------------------------------------------
__global__ __launch_bounds__(256) void gates(
    const float* __restrict__ c, const float* __restrict__ Ak,
    const float* __restrict__ abias, const float* __restrict__ xm,
    const float* __restrict__ hrec, const float* __restrict__ h,
    float* __restrict__ out)
{
    __shared__ float cs[Ff];
    const int b = blockIdx.y;
    const int u = blockIdx.x * 256 + threadIdx.x;
    cs[threadIdx.x]       = c[b * Uu + threadIdx.x];
    cs[threadIdx.x + 256] = c[b * Uu + threadIdx.x + 256];
    __syncthreads();

    float az = abias[u], ar = abias[Uu + u], ah = abias[2 * Uu + u];
    for (int k = 0; k < Ff; ++k) {
        float cv = cs[k];
        const float* row = Ak + (size_t)k * U3;
        az += cv * row[u];
        ar += cv * row[Uu + u];
        ah += cv * row[2 * Uu + u];
    }
    const float* xb = xm + (size_t)b * U3;
    const float* rb = hrec + (size_t)b * U3;
    float z  = hsig(xb[u] + rb[u] + az);
    float r  = hsig(xb[Uu + u] + rb[Uu + u] + ar);
    float hh = tanhf(xb[2 * Uu + u] + r * rb[2 * Uu + u] + ah);
    float hv = h[b * Uu + u];
    out[b * Uu + u] = z * hv + (1.f - z) * hh;
}

// ---------------------------------------------------------------------------
extern "C" void kernel_launch(void* const* d_in, const int* in_sizes, int n_in,
                              void* d_out, int out_size, void* d_ws, size_t ws_size,
                              hipStream_t stream) {
    const float* x     = (const float*)d_in[0];
    const float* h     = (const float*)d_in[1];
    const float* ann   = (const float*)d_in[2];
    const float* kern  = (const float*)d_in[3];
    const float* rker  = (const float*)d_in[4];
    const float* Ak    = (const float*)d_in[5];
    const float* Wa    = (const float*)d_in[6];
    const float* Ua    = (const float*)d_in[7];
    const float* Va    = (const float*)d_in[8];
    const float* bias  = (const float*)d_in[9];
    const float* abias = (const float*)d_in[10];
    const float* Wab   = (const float*)d_in[11];
    const float* Uab   = (const float*)d_in[12];
    float* out = (float*)d_out;

    // Workspace carve-up: 786,432 B fp32 scratch + 512 KB bf16 UaT.
    float* ws     = (float*)d_ws;
    float* hwa    = ws;                       // 32*512
    float* xm     = hwa  + Bb * Uu;           // 32*1536
    float* hrec   = xm   + Bb * U3;           // 32*1536
    float* scores = hrec + Bb * U3;           // 32*2048
    float* ctx    = scores + Bb * Tt;         // 32*512
    unsigned short* UaT = (unsigned short*)(ctx + Bb * Uu);  // 512*512 bf16
    (void)ws_size; (void)in_sizes; (void)n_in; (void)out_size;

    ua_transpose_bf16<<<dim3(Uu / 32, Ff / 32), 256, 0, stream>>>(Ua, UaT);

    small_mm<<<dim3((Bb * (Uu + U3 + U3) + 255) / 256), 256, 0, stream>>>(
        x, h, kern, rker, Wa, bias, Wab, Uab, hwa, xm, hrec);

    hipMemsetAsync(scores, 0, (size_t)Bb * Tt * sizeof(float), stream);

    attn_scores_wmma<<<dim3(Tt / A_ROWS, Bb, Uu / BN), 256, 0, stream>>>(
        ann, UaT, hwa, Va, scores);

    softmax_ctx<<<dim3(Bb, Ff / 256), 256, 0, stream>>>(scores, ann, ctx);

    gates<<<dim3(Uu / 256, Bb), 256, 0, stream>>>(
        ctx, Ak, abias, xm, hrec, h, out);
}
